// NeuralODEModel_46909632806974
// MI455X (gfx1250) — compile-verified, matrix operand order checked by reading, not run on hardware
//
#include <hip/hip_runtime.h>
#include <hip/hip_bf16.h>
#include <math.h>

typedef __attribute__((ext_vector_type(16))) __bf16 v16bf;
typedef __attribute__((ext_vector_type(8)))  __bf16 v8bf;
typedef __attribute__((ext_vector_type(8)))  float  v8f;

namespace {
constexpr int Bsz = 16384, Dd = 256, Hh = 512;
constexpr int ROWS    = 32;             // batch rows per workgroup (2 M-tiles)
constexpr int NTHREAD = 256;            // 8 wave32 -> 2 waves/SIMD -> ~512 VGPR/wave cap
constexpr int NSTEPS  = 16;
constexpr float DT    = 1.0f / 16.0f;

// padded LDS strides (bf16 elements) to spread banks
constexpr int ZSTR = Dd + 8;            // 264
constexpr int HSTR = Hh + 8;            // 520

// LDS byte layout
constexpr int YOFF = 0;                               // y      fp32 [32][256] = 32 KB
constexpr int KOFF = YOFF + ROWS * Dd * 4;            // k1..k6 fp32 6x32 KB   = 192 KB
constexpr int ZOFF = KOFF + 6 * ROWS * Dd * 4;        // z      bf16 [32][264] = 16.5 KB
constexpr int HOFF = ZOFF + ROWS * ZSTR * 2;          // h      bf16 [32][520] = 32.5 KB
constexpr int SMEM_BYTES = HOFF + ROWS * HSTR * 2;    // 279552 B < 320 KB/WGP

constexpr int KT1 = Dd / 32, NT1 = Hh / 16;           // GEMM1: 8 K-tiles, 32 N-tiles
constexpr int KT2 = Hh / 32, NT2 = Dd / 16;           // GEMM2: 16 K-tiles, 16 N-tiles
}

// dopri5 tableau
__constant__ float ATAB[6][5] = {
    {0.f, 0.f, 0.f, 0.f, 0.f},
    {1.f/5.f, 0.f, 0.f, 0.f, 0.f},
    {3.f/40.f, 9.f/40.f, 0.f, 0.f, 0.f},
    {44.f/45.f, -56.f/15.f, 32.f/9.f, 0.f, 0.f},
    {19372.f/6561.f, -25360.f/2187.f, 64448.f/6561.f, -212.f/729.f, 0.f},
    {9017.f/3168.f, -355.f/33.f, 46732.f/5247.f, 49.f/176.f, -5103.f/18656.f}};
__constant__ float BTAB[5] = {35.f/384.f, 500.f/1113.f, 125.f/192.f,
                              -2187.f/6784.f, 11.f/84.f};   // k1,k3,k4,k5,k6

// ---- fragment helpers -------------------------------------------------------
// A (16x32 bf16, row-major source): lane half selects K 0..7/16..23 vs 8..15/24..31
__device__ inline v16bf frag_a(const __bf16* base, int stride, int lane, int kt) {
    const int m  = lane & 15;
    const int hi = (lane >> 4) & 1;
    const __bf16* p = base + m * stride + kt * 32 + hi * 8;
    v8bf lo = *(const v8bf*)(p);          // K = 8*hi + 0..7
    v8bf hv = *(const v8bf*)(p + 16);     // K = 8*hi + 16..23
    v16bf r;
#pragma unroll
    for (int i = 0; i < 8; ++i) { r[i] = lo[i]; r[i + 8] = hv[i]; }
    return r;
}

// B fragments pre-swizzled by convert_weights: 32 B contiguous per lane per tile
__device__ inline v16bf frag_b(const __bf16* blob, int nt_total, int kt, int nt, int lane) {
    const __bf16* p = blob + (((size_t)(kt * nt_total + nt)) * 32 + lane) * 16;
    v8bf lo = *(const v8bf*)(p);
    v8bf hv = *(const v8bf*)(p + 8);
    v16bf r;
#pragma unroll
    for (int i = 0; i < 8; ++i) { r[i] = lo[i]; r[i + 8] = hv[i]; }
    return r;
}

__device__ inline float fast_tanh(float v) {
    float e = __expf(-2.0f * fabsf(v));
    float t = (1.0f - e) / (1.0f + e);
    return copysignf(t, v);
}

// ---- prologue: fp32 -> bf16 weights, pre-swizzled into B-fragment order -----
__global__ void convert_weights(const float* __restrict__ W1, const float* __restrict__ W2,
                                __bf16* __restrict__ w1b, __bf16* __restrict__ w2b) {
    const int NW = Dd * Hh;               // 131072 elements per matrix
    int d = blockIdx.x * blockDim.x + threadIdx.x;
    if (d < NW) {                         // W1: K=256, N=512
        int tt = d >> 9, rem = d & 511, l = rem >> 4, e = rem & 15;
        int kt = tt / NT1, nt = tt % NT1, hi = l >> 4;
        int k = kt * 32 + ((e < 8) ? (e + 8 * hi) : (e + 8 + 8 * hi));
        int n = nt * 16 + (l & 15);
        w1b[d] = (__bf16)W1[k * Hh + n];
    } else if (d < 2 * NW) {              // W2: K=512, N=256
        int d2 = d - NW;
        int tt = d2 >> 9, rem = d2 & 511, l = rem >> 4, e = rem & 15;
        int kt = tt / NT2, nt = tt % NT2, hi = l >> 4;
        int k = kt * 32 + ((e < 8) ? (e + 8 * hi) : (e + 8 + 8 * hi));
        int n = nt * 16 + (l & 15);
        w2b[d2] = (__bf16)W2[k * Dd + n];
    }
}

// ---- main: full dopri5 integration, one workgroup per 32-row tile -----------
// 8 waves/WG: W2 fragments register-resident (256 VGPR/lane) for all 96 stages;
// W1 fragments streamed from L2 each stage. An opaque integer offset (not an
// opaque pointer!) defeats LICM while keeping global address-space provenance,
// so the stream uses global_load_b128 and DScnt/LOADcnt waits stay decoupled.
__global__ __launch_bounds__(NTHREAD) void node_kernel(
    const float* __restrict__ x,
    const __bf16* __restrict__ W1b, const float* __restrict__ b1,
    const __bf16* __restrict__ W2b, const float* __restrict__ b2,
    const float* __restrict__ Wfc, const float* __restrict__ bfc,
    float* __restrict__ out) {
    extern __shared__ char smem[];
    float*  sY = (float*)(smem + YOFF);
    float*  sK = (float*)(smem + KOFF);
    __bf16* sZ = (__bf16*)(smem + ZOFF);
    __bf16* sH = (__bf16*)(smem + HOFF);

    const int tid  = threadIdx.x;
    const int wave = tid >> 5, lane = tid & 31;   // wave in [0,8)
    const int cl   = lane & 15, hi = (lane >> 4) & 1;
    const int row0 = blockIdx.x * ROWS;

    // ---- register-resident W2 fragments (32 frags = 256 VGPR, reused 96x) ---
    v16bf bw2[KT2][2];                    // GEMM2: nt = wave, wave+8
#pragma unroll
    for (int kt = 0; kt < KT2; ++kt)
#pragma unroll
        for (int nj = 0; nj < 2; ++nj)
            bw2[kt][nj] = frag_b(W2b, NT2, kt, wave + nj * 8, lane);

    // register-resident bias values for owned columns
    float bn1[4], bn2[2];
#pragma unroll
    for (int nj = 0; nj < 4; ++nj) bn1[nj] = b1[(wave + nj * 8) * 16 + cl];
#pragma unroll
    for (int nj = 0; nj < 2; ++nj) bn2[nj] = b2[(wave + nj * 8) * 16 + cl];

    // load y0 tile
    for (int i = tid; i < ROWS * Dd; i += NTHREAD)
        sY[i] = x[(size_t)row0 * Dd + i];
    __syncthreads();

    for (int step = 0; step < NSTEPS; ++step) {
        for (int s = 0; s < 6; ++s) {
            // z = y + dt * sum_j a_sj k_j   (fp32 math, round to bf16 once)
            for (int i = tid; i < ROWS * Dd; i += NTHREAD) {
                float z = sY[i];
                for (int j = 0; j < s; ++j)
                    z += DT * ATAB[s][j] * sK[j * ROWS * Dd + i];
                sZ[(i >> 8) * ZSTR + (i & 255)] = (__bf16)z;
            }
            __syncthreads();

            // GEMM1: h = tanh(z @ W1 + b1)   [32,256]x[256,512]
            {
                // loop-variant opaque zero offset: blocks hoisting of the W1
                // fragment loads out of the stage loop, keeps provenance.
                int zoff = 0;
                asm volatile("" : "+s"(zoff));
                const __bf16* w1p = W1b + zoff;

                v8f acc[2][4] = {};
#pragma unroll
                for (int kt = 0; kt < KT1; ++kt) {
                    v16bf a0 = frag_a(sZ,             ZSTR, lane, kt);
                    v16bf a1 = frag_a(sZ + 16 * ZSTR, ZSTR, lane, kt);
#pragma unroll
                    for (int nj = 0; nj < 4; ++nj) {
                        v16bf b = frag_b(w1p, NT1, kt, wave + nj * 8, lane);
                        acc[0][nj] = __builtin_amdgcn_wmma_f32_16x16x32_bf16(
                            false, a0, false, b, (short)0, acc[0][nj], false, false);
                        acc[1][nj] = __builtin_amdgcn_wmma_f32_16x16x32_bf16(
                            false, a1, false, b, (short)0, acc[1][nj], false, false);
                    }
                }
#pragma unroll
                for (int rt = 0; rt < 2; ++rt)
#pragma unroll
                    for (int nj = 0; nj < 4; ++nj) {
                        int col = (wave + nj * 8) * 16 + cl;
#pragma unroll
                        for (int r = 0; r < 8; ++r) {
                            int row = rt * 16 + r + 8 * hi;
                            sH[row * HSTR + col] =
                                (__bf16)fast_tanh(acc[rt][nj][r] + bn1[nj]);
                        }
                    }
            }
            __syncthreads();

            // GEMM2: k_s = h @ W2 + b2       [32,512]x[512,256]
            {
                v8f acc[2][2] = {};
#pragma unroll
                for (int kt = 0; kt < KT2; ++kt) {
                    v16bf a0 = frag_a(sH,             HSTR, lane, kt);
                    v16bf a1 = frag_a(sH + 16 * HSTR, HSTR, lane, kt);
#pragma unroll
                    for (int nj = 0; nj < 2; ++nj) {
                        acc[0][nj] = __builtin_amdgcn_wmma_f32_16x16x32_bf16(
                            false, a0, false, bw2[kt][nj], (short)0, acc[0][nj], false, false);
                        acc[1][nj] = __builtin_amdgcn_wmma_f32_16x16x32_bf16(
                            false, a1, false, bw2[kt][nj], (short)0, acc[1][nj], false, false);
                    }
                }
                float* sKs = sK + s * ROWS * Dd;
#pragma unroll
                for (int rt = 0; rt < 2; ++rt)
#pragma unroll
                    for (int nj = 0; nj < 2; ++nj) {
                        int col = (wave + nj * 8) * 16 + cl;
#pragma unroll
                        for (int r = 0; r < 8; ++r) {
                            int row = rt * 16 + r + 8 * hi;
                            sKs[row * Dd + col] = acc[rt][nj][r] + bn2[nj];
                        }
                    }
            }
            __syncthreads();
        }

        // y += dt * (B1 k1 + B3 k3 + B4 k4 + B5 k5 + B6 k6)
        for (int i = tid; i < ROWS * Dd; i += NTHREAD) {
            float acc = BTAB[0] * sK[i]
                      + BTAB[1] * sK[2 * ROWS * Dd + i]
                      + BTAB[2] * sK[3 * ROWS * Dd + i]
                      + BTAB[3] * sK[4 * ROWS * Dd + i]
                      + BTAB[4] * sK[5 * ROWS * Dd + i];
            sY[i] += DT * acc;
        }
        __syncthreads();
    }

    // out = y1 @ Wfc + bfc  (tiny: one row per thread)
    if (tid < ROWS) {
        float a = bfc[0];
        for (int j = 0; j < Dd; ++j) a += sY[tid * Dd + j] * Wfc[j];
        out[row0 + tid] = a;
    }
}

extern "C" void kernel_launch(void* const* d_in, const int* in_sizes, int n_in,
                              void* d_out, int out_size, void* d_ws, size_t ws_size,
                              hipStream_t stream) {
    const float* x   = (const float*)d_in[0];
    const float* W1  = (const float*)d_in[1];
    const float* b1  = (const float*)d_in[2];
    const float* W2  = (const float*)d_in[3];
    const float* b2  = (const float*)d_in[4];
    const float* Wfc = (const float*)d_in[5];
    const float* bfc = (const float*)d_in[6];
    float* out = (float*)d_out;

    __bf16* w1b = (__bf16*)d_ws;              // 256 KB
    __bf16* w2b = w1b + (size_t)Dd * Hh;      // 256 KB

    convert_weights<<<(2 * Dd * Hh + 255) / 256, 256, 0, stream>>>(W1, W2, w1b, w2b);
    node_kernel<<<Bsz / ROWS, NTHREAD, SMEM_BYTES, stream>>>(x, w1b, b1, w2b, b2,
                                                             Wfc, bfc, out);
}
